// SelfAttention_33423435498417
// MI455X (gfx1250) — matrix operand order repaired and checked by reading, not compiled, not run on hardware
//
#include <hip/hip_runtime.h>
#include <hip/hip_bf16.h>
#include <cstdint>

typedef __attribute__((ext_vector_type(16))) __bf16 v16bf;
typedef __attribute__((ext_vector_type(8)))  __bf16 v8bf;
typedef __attribute__((ext_vector_type(8)))  float  v8f;
typedef __attribute__((ext_vector_type(4)))  unsigned int v4u;
typedef __attribute__((ext_vector_type(8)))  int          v8i;
typedef __attribute__((ext_vector_type(4)))  int          v4i;

#define WMMA_BF16(a, b, c) \
    __builtin_amdgcn_wmma_f32_16x16x32_bf16(false, (a), false, (b), (short)0, (c), false, false)

#if __has_builtin(__builtin_amdgcn_tensor_load_to_lds) && \
    __has_builtin(__builtin_amdgcn_s_wait_tensorcnt)
#define HAVE_TDM 1
#else
#define HAVE_TDM 0
#endif

static constexpr int D  = 512;
static constexpr int S  = 2048;
static constexpr int BB = 4;
static constexpr int LROW = 520;   // padded LDS row stride (512 elems + 16B pad)

__device__ inline v8bf ld8bf(const __bf16* p) { return *(const v8bf*)p; }

__device__ inline v16bf join8(v8bf lo, v8bf hi) {
    return __builtin_shufflevector(lo, hi, 0,1,2,3,4,5,6,7,8,9,10,11,12,13,14,15);
}

#if HAVE_TDM
// ---------------------------------------------------------------------------
// TDM: DMA `rows` x 512 bf16 tile (row stride 512 elems in memory) into LDS
// with a 16-byte pad after every 1024-byte row (LDS row stride = 520 elems).
// D# per CDNA5 ISA 8.3/8.4: pad_interval=7 (256 DW), pad_amount=3 (4 DW).
// Issue from ONE wave; caller waits TENSORcnt then barriers.
// ---------------------------------------------------------------------------
__device__ inline void tdm_load_rows_512bf16(const __bf16* gsrc, __bf16* ldst, int rows) {
    uint64_t ga = (uint64_t)(uintptr_t)gsrc;
    uint32_t la = (uint32_t)(uintptr_t)ldst;           // low 32 bits = LDS offset
    v4u g0;
    g0[0] = 1u;                                         // count=1, user descriptor
    g0[1] = la;                                         // lds_addr
    g0[2] = (uint32_t)(ga & 0xFFFFFFFFu);               // global_addr[31:0]
    g0[3] = (uint32_t)((ga >> 32) & 0x01FFFFFFu) | (2u << 30);  // addr[56:32], type=2
    v8i g1;
    g1[0] = (int)((1u << 16) |                          // data_size = 2 bytes
                  (1u << 20) |                          // pad_enable
                  (7u << 22) |                          // pad_interval: 256 DWORDs
                  (3u << 25));                          // pad_amount:   4 DWORDs
    g1[1] = (int)(512u << 16);                          // tensor_dim0 = 512 (lo16)
    g1[2] = (int)((uint32_t)rows << 16);                // tensor_dim0 hi=0 | tensor_dim1 lo = rows
    g1[3] = (int)(512u << 16);                          // tensor_dim1 hi=0 | tile_dim0 = 512
    g1[4] = (int)(rows & 0xFFFF);                       // tile_dim1 = rows, tile_dim2 = 0
    g1[5] = (int)512;                                   // tensor_dim0_stride[31:0] = 512
    g1[6] = 0;                                          // stride hi | tensor_dim1_stride lo
    g1[7] = 0;
    v4i z4 = {0, 0, 0, 0};
    v8i z8 = {0, 0, 0, 0, 0, 0, 0, 0};
    __builtin_amdgcn_tensor_load_to_lds(g0, g1, z4, z4, z8, 0);
}
#else
// Fallback: cooperative copy through VGPRs (all 128 threads participate).
__device__ inline void coop_load_rows_512bf16(const __bf16* gsrc, __bf16* ldst, int rows) {
    int total = rows * 64;                              // 16B chunks
    for (int idx = threadIdx.x; idx < total; idx += 128) {
        int r = idx >> 6, cc = (idx & 63) * 8;
        *(v8bf*)(ldst + r * LROW + cc) = *(const v8bf*)(gsrc + r * 512 + cc);
    }
}
#endif

// ---------------------------------------------------------------------------
// Kernel 0a: convert x (fp32) -> xb (bf16), flat
// ---------------------------------------------------------------------------
__global__ __launch_bounds__(256) void cvt_x_kernel(const float* __restrict__ x,
                                                    __bf16* __restrict__ xb, int n) {
    int i = blockIdx.x * 256 + threadIdx.x;
    if (i < n) xb[i] = (__bf16)x[i];
}

// ---------------------------------------------------------------------------
// Kernel 0b: W[d][e] fp32 -> Wt[e][d] bf16 (transpose+convert)
// ---------------------------------------------------------------------------
__global__ __launch_bounds__(256) void transpose_w_kernel(const float* __restrict__ W,
                                                          __bf16* __restrict__ Wt) {
    int i = blockIdx.x * 256 + threadIdx.x;   // i = e*D + d
    int e = i >> 9;
    int d = i & (D - 1);
    Wt[i] = (__bf16)W[d * D + e];
}

// ---------------------------------------------------------------------------
// Kernel 1: QKV projection GEMM (bf16 WMMA, fp32 accum).
//   Block = 4 waves sharing one 16-row x-tile (TDM-staged in LDS); each wave
//   owns a 64-col strip of one output matrix.  Q pre-scaled by 1/sqrt(D).
//   V stored transposed: Vt[b][e][s].
// ---------------------------------------------------------------------------
__global__ __launch_bounds__(128) void qkv_gemm_kernel(
    const __bf16* __restrict__ xb,
    const __bf16* __restrict__ Wtq, const __bf16* __restrict__ Wtk, const __bf16* __restrict__ Wtv,
    const float*  __restrict__ bq,  const float*  __restrict__ bk,  const float*  __restrict__ bv,
    __bf16* __restrict__ Qb, __bf16* __restrict__ Kb, __bf16* __restrict__ Vt)
{
    __shared__ __bf16 alds[16 * LROW];      // shared 16 x 512 x-tile

    const int wave = threadIdx.x >> 5;
    const int lane = threadIdx.x & 31;
    const int h    = lane >> 4;
    const int c    = lane & 15;

    int task   = blockIdx.x * 4 + wave;     // 12288 tasks; block's 4 share mtile+mat
    int mat    = task / 4096;               // 0=Q 1=K 2=V
    int rem    = task - mat * 4096;
    int mtile  = rem >> 3;
    int nbase  = (rem & 7) * 64;
    int row0   = mtile * 16;

    const __bf16* Wt   = (mat == 0) ? Wtq : (mat == 1) ? Wtk : Wtv;
    const float*  bias = (mat == 0) ? bq  : (mat == 1) ? bk  : bv;

#if HAVE_TDM
    if (wave == 0) {
        tdm_load_rows_512bf16(xb + (size_t)row0 * D, alds, 16);
        __builtin_amdgcn_s_wait_tensorcnt(0);
    }
#else
    coop_load_rows_512bf16(xb + (size_t)row0 * D, alds, 16);
#endif
    __syncthreads();

    v8f acc[4];
    #pragma unroll
    for (int j = 0; j < 4; ++j) acc[j] = (v8f)0.0f;

    const __bf16* brow[4];
    #pragma unroll
    for (int j = 0; j < 4; ++j) brow[j] = Wt + (size_t)(nbase + j * 16 + c) * D;

    for (int k0 = 0; k0 < D; k0 += 32) {
        // A 16x32 frag from LDS: row = c, k-octets selected by h
        v16bf a = join8(*(const v8bf*)(alds + c * LROW + k0 + h * 16),
                        *(const v8bf*)(alds + c * LROW + k0 + h * 16 + 8));
        #pragma unroll
        for (int j = 0; j < 4; ++j) {
            // B 32x16 frag: elem i -> k = k0 + h*16 + i, col = nbase+j*16+c
            v16bf b = join8(ld8bf(brow[j] + k0 + h * 16),
                            ld8bf(brow[j] + k0 + h * 16 + 8));
            acc[j] = WMMA_BF16(a, b, acc[j]);
        }
    }

    if (mat < 2) {
        const float scale = (mat == 0) ? 0.044194173824159216f /* 1/sqrt(512) */ : 1.0f;
        __bf16* Out = (mat == 0) ? Qb : Kb;
        #pragma unroll
        for (int j = 0; j < 4; ++j) {
            float bv_ = bias[nbase + j * 16 + c];
            #pragma unroll
            for (int r = 0; r < 8; ++r) {                 // C row = r + 8h
                int rr = row0 + r + 8 * h;
                Out[(size_t)rr * D + nbase + j * 16 + c] = (__bf16)((acc[j][r] + bv_) * scale);
            }
        }
    } else {
        int b0 = row0 / S;
        int s0 = row0 - b0 * S;
        #pragma unroll
        for (int j = 0; j < 4; ++j) {
            int e = nbase + j * 16 + c;
            float bv_ = bias[e];
            v8bf pk;
            #pragma unroll
            for (int r = 0; r < 8; ++r) pk[r] = (__bf16)(acc[j][r] + bv_);
            *(v8bf*)(Vt + ((size_t)b0 * D + e) * S + s0 + 8 * h) = pk;
        }
    }
}

// ---------------------------------------------------------------------------
// Kernel 2: flash attention.  Block = 1 q-tile (16 queries) = 4 waves; wave w
// owns output dims [w*128, w*128+128).  Q tile + double-buffered 32-key K
// tiles staged in LDS via TDM; TDM of tile t+1 overlaps WMMAs of tile t.
// ---------------------------------------------------------------------------
__global__ __launch_bounds__(128) void attn_kernel(
    const __bf16* __restrict__ Qb, const __bf16* __restrict__ Kb,
    const __bf16* __restrict__ Vt, float* __restrict__ out)
{
    __shared__ __bf16 qlds[16 * LROW];        // Q tile 16 x 512
    __shared__ __bf16 klds[2][32 * LROW];     // double-buffered K tile 32 x 512
    __shared__ __bf16 plds[4][16 * 40];       // per-wave P tile 16 x 32 (stride 40)

    const int wave = threadIdx.x >> 5;
    const int lane = threadIdx.x & 31;
    const int h    = lane >> 4;
    const int c    = lane & 15;

    const int bb = blockIdx.x >> 7;
    const int q0 = (blockIdx.x & 127) * 16;
    const int d0 = wave * 128;

    const __bf16* Kbat = Kb + (size_t)bb * S * D;
    const __bf16* Vbat = Vt + (size_t)bb * D * S;

#if HAVE_TDM
    if (wave == 0) {
        tdm_load_rows_512bf16(Qb + ((size_t)bb * S + q0) * D, qlds, 16);
        tdm_load_rows_512bf16(Kbat, klds[0], 32);
        __builtin_amdgcn_s_wait_tensorcnt(0);
    }
#else
    coop_load_rows_512bf16(Qb + ((size_t)bb * S + q0) * D, qlds, 16);
    coop_load_rows_512bf16(Kbat, klds[0], 32);
#endif
    __syncthreads();

    v8f acc[8];
    #pragma unroll
    for (int j = 0; j < 8; ++j) acc[j] = (v8f)0.0f;
    float mrow[8], lrow[8];
    #pragma unroll
    for (int r = 0; r < 8; ++r) { mrow[r] = -1e30f; lrow[r] = 0.0f; }

    __bf16* myp = plds[wave];
    int cur = 0;

    #pragma unroll 1
    for (int kt = 0; kt < S; kt += 32) {
        const __bf16* kcur = klds[cur];
#if HAVE_TDM
        if (wave == 0 && kt + 32 < S)    // prefetch next K tile into other buffer
            tdm_load_rows_512bf16(Kbat + (size_t)(kt + 32) * D, klds[cur ^ 1], 32);
#endif
        // ---- scores: S_tile[16 x 32] = Q . K^T (pre-scaled by 1/sqrt(D)) ----
        v8f s0 = (v8f)0.0f, s1 = (v8f)0.0f;
        #pragma unroll
        for (int kk = 0; kk < 16; ++kk) {
            v16bf a = join8(*(const v8bf*)(qlds + c * LROW + kk * 32 + h * 16),
                            *(const v8bf*)(qlds + c * LROW + kk * 32 + h * 16 + 8));
            v16bf b0 = join8(*(const v8bf*)(kcur + c * LROW + kk * 32 + h * 16),
                             *(const v8bf*)(kcur + c * LROW + kk * 32 + h * 16 + 8));
            s0 = WMMA_BF16(a, b0, s0);
            v16bf b1 = join8(*(const v8bf*)(kcur + (16 + c) * LROW + kk * 32 + h * 16),
                             *(const v8bf*)(kcur + (16 + c) * LROW + kk * 32 + h * 16 + 8));
            s1 = WMMA_BF16(a, b1, s1);
        }

        // ---- online softmax (row r+8h lives in one 16-lane half) ----
        float palpha[8];
        #pragma unroll
        for (int r = 0; r < 8; ++r) {
            float tmax = fmaxf(s0[r], s1[r]);
            #pragma unroll
            for (int m = 1; m < 16; m <<= 1)
                tmax = fmaxf(tmax, __shfl_xor(tmax, m, 32));
            float mnew = fmaxf(mrow[r], tmax);
            float al   = __expf(mrow[r] - mnew);
            float p0   = __expf(s0[r] - mnew);
            float p1   = __expf(s1[r] - mnew);
            float rs   = p0 + p1;
            #pragma unroll
            for (int m = 1; m < 16; m <<= 1)
                rs += __shfl_xor(rs, m, 32);
            mrow[r]   = mnew;
            lrow[r]   = lrow[r] * al + rs;
            palpha[r] = al;
            myp[(r + 8 * h) * 40 + c]      = (__bf16)p0;   // P (D-layout) -> LDS
            myp[(r + 8 * h) * 40 + 16 + c] = (__bf16)p1;
        }
        #pragma unroll
        for (int j = 0; j < 8; ++j)
            #pragma unroll
            for (int r = 0; r < 8; ++r) acc[j][r] *= palpha[r];

        asm volatile("s_wait_dscnt 0" ::: "memory");   // in-wave LDS transpose fence

        // ---- re-read P in A-fragment layout ----
        v16bf pa = join8(*(const v8bf*)(myp + c * 40 + h * 16),
                         *(const v8bf*)(myp + c * 40 + h * 16 + 8));

        // ---- acc += P . V (B frags contiguous thanks to transposed V) ----
        const __bf16* vcol = Vbat + (size_t)(d0 + c) * S + kt + h * 16;
        if (kt + 32 < S) __builtin_prefetch((const void*)(vcol + 32), 0, 0);
        #pragma unroll
        for (int j = 0; j < 8; ++j) {
            const __bf16* vp = vcol + (size_t)(j * 16) * S;
            v16bf vb = join8(ld8bf(vp), ld8bf(vp + 8));
            acc[j] = WMMA_BF16(pa, vb, acc[j]);
        }

#if HAVE_TDM
        if (wave == 0) __builtin_amdgcn_s_wait_tensorcnt(0);
#else
        if (kt + 32 < S)
            coop_load_rows_512bf16(Kbat + (size_t)(kt + 32) * D, klds[cur ^ 1], 32);
#endif
        __syncthreads();
        cur ^= 1;
    }

    // ---- epilogue: normalize and store fp32 ----
    #pragma unroll
    for (int j = 0; j < 8; ++j) {
        #pragma unroll
        for (int r = 0; r < 8; ++r) {
            int qq = q0 + r + 8 * h;
            out[((size_t)bb * S + qq) * D + d0 + j * 16 + c] = acc[j][r] / lrow[r];
        }
    }
}

// ---------------------------------------------------------------------------
extern "C" void kernel_launch(void* const* d_in, const int* in_sizes, int n_in,
                              void* d_out, int out_size, void* d_ws, size_t ws_size,
                              hipStream_t stream) {
    const float* x  = (const float*)d_in[0];
    const float* Wq = (const float*)d_in[1];
    const float* bq = (const float*)d_in[2];
    const float* Wk = (const float*)d_in[3];
    const float* bk = (const float*)d_in[4];
    const float* Wv = (const float*)d_in[5];
    const float* bv = (const float*)d_in[6];

    const size_t nTok = (size_t)BB * S;
    const size_t nXD  = nTok * D;

    __bf16* Qb  = (__bf16*)d_ws;
    __bf16* Kb  = Qb  + nXD;
    __bf16* Vt  = Kb  + nXD;                 // transposed [B][D][S]
    __bf16* xb  = Vt  + nXD;
    __bf16* Wtq = xb  + nXD;                 // [e][d]
    __bf16* Wtk = Wtq + (size_t)D * D;
    __bf16* Wtv = Wtk + (size_t)D * D;

    cvt_x_kernel<<<(int)(nXD / 256), 256, 0, stream>>>(x, xb, (int)nXD);
    transpose_w_kernel<<<(D * D) / 256, 256, 0, stream>>>(Wq, Wtq);
    transpose_w_kernel<<<(D * D) / 256, 256, 0, stream>>>(Wk, Wtk);
    transpose_w_kernel<<<(D * D) / 256, 256, 0, stream>>>(Wv, Wtv);

    qkv_gemm_kernel<<<3072, 128, 0, stream>>>(xb, Wtq, Wtk, Wtv, bq, bk, bv, Qb, Kb, Vt);
    attn_kernel<<<512, 128, 0, stream>>>(Qb, Kb, Vt, (float*)d_out);
}
